// Group_Prime_Kernel_FFN_53901839565655
// MI455X (gfx1250) — compile-verified, matrix operand order checked by reading, not run on hardware
//
#include <hip/hip_runtime.h>

// CDNA5 / gfx1250 fused ConvNeXt-style block:
//   LN(ch) -> pw 64->256 -> 4x { dwconv k, pw 64x64, dwconv gate, mul } -> pw 256->64 -> *scale + x
// Pointwise GEMMs on v_wmma_f32_16x16x32_f16 (f16 in, f32 acc). A-fragments are
// register-cached per row-tile; depthwise convs are interior-specialized and
// compute two time positions per thread (shared tap loads, packed f32 FMA).

typedef __attribute__((ext_vector_type(16))) _Float16 v16h;
typedef __attribute__((ext_vector_type(8)))  float    v8f;

#define C_   64
#define TILE 64
#define HALO 16
#define TP   96          // TILE + 2*HALO
#define TT   48000
#define NB   8

struct Params {
  const float *x, *ln_w, *ln_b, *pf_w, *pf_b, *pl_w, *pl_b, *scale;
  const float *attn_dw[4], *attn_db[4], *attn_pw[4], *attn_pb[4];
  const float *conv_dw[4], *conv_db[4];
  float* out;
};

// K-pair placement inside a 16-bit A/B fragment (ISA 7.12.2, 16-bit A 16x32):
// VGPR j (j<4): K = 2j + 8*half ; VGPR j>=4: K = 16 + 2(j-4) + 8*half
__device__ __forceinline__ int kmap(int j, int half) {
  return (j < 4 ? 2 * j : 16 + 2 * (j - 4)) + 8 * half;
}

// A fragment (16 rows x 32 K) from row-major f32 matrix W (leading dim ldw), cvt to f16.
__device__ __forceinline__ v16h load_a_f32(const float* __restrict__ W, int ldw,
                                           int row0, int k0, int lane) {
  const int r = row0 + (lane & 15), h = lane >> 4;
  v16h a;
#pragma unroll
  for (int j = 0; j < 8; ++j) {
    const int k = k0 + kmap(j, h);
    a[2 * j]     = (_Float16)W[r * ldw + k];
    a[2 * j + 1] = (_Float16)W[r * ldw + k + 1];
  }
  return a;
}

// B fragment (32 K x 16 cols) from f16 LDS stored [n][k], K contiguous (ld = 64).
__device__ __forceinline__ v16h load_b_t(const _Float16* Bt, int n0, int k0, int lane) {
  const int n = n0 + (lane & 15), h = lane >> 4;
  v16h b;
#pragma unroll
  for (int j = 0; j < 8; ++j) {
    const int k = k0 + kmap(j, h);
    b[2 * j]     = Bt[n * 64 + k];
    b[2 * j + 1] = Bt[n * 64 + k + 1];
  }
  return b;
}

__global__ __launch_bounds__(256) void fused_block_kernel(Params p) {
  extern __shared__ char smem[];
  float*    xs   = (float*)smem;                    // [c][q] f32, 64*96*4   = 24576
  float*    gate = (float*)(smem + 24576);          // [t][c] f32, 64*64*4   = 16384
  _Float16* ys   = (_Float16*)(smem + 40960);       // [q][c] f16, 96*64*2   = 12288
  _Float16* hs   = (_Float16*)(smem + 53248);       // [ch][q] f16, 256*96*2 = 49152
  _Float16* dwa  = (_Float16*)(smem + 102400);      // [t][c] f16, 64*64*2   = 8192
  _Float16* oi   = (_Float16*)(smem + 110592);      // [t][c] f16, 64*64*2   = 8192
                                                    // total = 118784 B (< 320 KB WGP LDS)
  const int tid  = threadIdx.x;
  const int lane = tid & 31, wv = tid >> 5;
  const int l15  = lane & 15, lh = lane >> 4;
  const int b    = blockIdx.x / (TT / TILE);
  const int t0   = (blockIdx.x % (TT / TILE)) * TILE;
  const float* __restrict__ xb = p.x + (size_t)b * C_ * TT;
  const bool interior = (t0 >= HALO) && (t0 + TILE + HALO <= TT);

  // ---- load x tile (with halo, zero-padded at sequence edges) ----
  if (interior) {
    for (int i = tid; i < C_ * TP; i += 256) {
      const int c = i / TP, q = i % TP;
      xs[c * TP + q] = xb[c * TT + (t0 - HALO + q)];
    }
  } else {
    for (int i = tid; i < C_ * TP; i += 256) {
      const int c = i / TP, q = i % TP;
      const int tg = t0 - HALO + q;
      xs[c * TP + q] = (tg >= 0 && tg < TT) ? xb[c * TT + tg] : 0.f;
    }
  }
  __syncthreads();

  // ---- channel LayerNorm per position, emit f16 [q][c] ----
  for (int q = tid; q < TP; q += 256) {
    float s = 0.f, s2 = 0.f;
    for (int c = 0; c < C_; ++c) { const float v = xs[c * TP + q]; s += v; s2 += v * v; }
    const float mu = s * (1.f / C_);
    const float var = s2 * (1.f / C_) - mu * mu;
    const float ri = rsqrtf(var + 1e-6f);
    for (int c = 0; c < C_; ++c) {
      const float v = (xs[c * TP + q] - mu) * ri * p.ln_w[c] + p.ln_b[c];
      ys[q * C_ + c] = (_Float16)v;
    }
  }
  __syncthreads();

  // ---- GEMM1: hs(256 x 96) = pf_w(256x64) @ y(64x96) + pf_b ----
  // Wave w owns row-tiles {2w, 2w+1}; A fragments register-cached across 6 col-tiles.
#pragma unroll
  for (int rr = 0; rr < 2; ++rr) {
    const int rt = 2 * wv + rr;
    const v16h a0 = load_a_f32(p.pf_w, 64, rt * 16, 0, lane);
    const v16h a1 = load_a_f32(p.pf_w, 64, rt * 16, 32, lane);
#pragma unroll
    for (int ct = 0; ct < 6; ++ct) {
      v8f d = {};
      d = __builtin_amdgcn_wmma_f32_16x16x32_f16(false, a0, false,
            load_b_t(ys, ct * 16, 0, lane), (short)0, d, false, false);
      d = __builtin_amdgcn_wmma_f32_16x16x32_f16(false, a1, false,
            load_b_t(ys, ct * 16, 32, lane), (short)0, d, false, false);
      const int n = ct * 16 + l15;
#pragma unroll
      for (int j = 0; j < 8; ++j) {
        const int m = rt * 16 + j + 8 * lh;
        hs[m * TP + n] = (_Float16)(d[j] + p.pf_b[m]);
      }
    }
  }
  __syncthreads();

  // ---- branches: dwconv -> attn GEMM -> gate mul -> accumulate pl GEMM ----
  // Output-tile ownership (64x64): rt = wv&3, ct in {2*(wv>>2), 2*(wv>>2)+1}.
  const int ort = wv & 3, oc0 = (wv >> 2) * 2;
  v8f acc[2] = {v8f{}, v8f{}};
  const int KS[4] = {3, 11, 23, 31};
#pragma unroll
  for (int br = 0; br < 4; ++br) {
    const int kk = KS[br], pad = (kk - 1) >> 1;
    const float* __restrict__ adw = p.attn_dw[br];
    const float* __restrict__ adb = p.attn_db[br];
    const float* __restrict__ cdw = p.conv_dw[br];
    const float* __restrict__ cdb = p.conv_db[br];
    const _Float16* hbr = hs + br * C_ * TP;

    // depthwise convs: two adjacent time positions per thread, shared tap loads
    for (int i = tid; i < C_ * (TILE / 2); i += 256) {
      const int c = i & 63, t = (i >> 6) * 2;
      const int base = c * TP + HALO + t - pad;
      float sa0 = 0.f, sa1 = 0.f, sg0 = 0.f, sg1 = 0.f;
      if (interior) {
        float h0 = (float)hbr[base];
#pragma unroll
        for (int j = 0; j < kk; ++j) {
          const float wa = adw[c * kk + j], wg = cdw[c * kk + j];
          const float h1 = (float)hbr[base + j + 1];
          sa0 += wa * h0; sa1 += wa * h1;     // packed-f32 FMA candidates
          sg0 += wg * h0; sg1 += wg * h1;
          h0 = h1;
        }
      } else {
#pragma unroll
        for (int j = 0; j < kk; ++j) {
          const float wa = adw[c * kk + j], wg = cdw[c * kk + j];
          const int tg0 = t0 + t + j - pad;
          if (tg0 >= 0 && tg0 < TT) {
            const float h0 = (float)hbr[base + j];
            sa0 += wa * h0; sg0 += wg * h0;
          }
          if (tg0 + 1 >= 0 && tg0 + 1 < TT) {
            const float h1 = (float)hbr[base + j + 1];
            sa1 += wa * h1; sg1 += wg * h1;
          }
        }
      }
      const float ab = adb[c], cb = cdb[c];
      dwa[t * C_ + c]        = (_Float16)(sa0 + ab);
      dwa[(t + 1) * C_ + c]  = (_Float16)(sa1 + ab);
      gate[t * C_ + c]       = sg0 + cb;
      gate[(t + 1) * C_ + c] = sg1 + cb;
    }
    __syncthreads();

    // GEMM2: attn = attn_pw(64x64) @ dwa ; then oi = (attn + b) * gate
    {
      const v16h a0 = load_a_f32(p.attn_pw[br], 64, ort * 16, 0, lane);
      const v16h a1 = load_a_f32(p.attn_pw[br], 64, ort * 16, 32, lane);
#pragma unroll
      for (int cc = 0; cc < 2; ++cc) {
        const int ct = oc0 + cc;
        v8f d = {};
        d = __builtin_amdgcn_wmma_f32_16x16x32_f16(false, a0, false,
              load_b_t(dwa, ct * 16, 0, lane), (short)0, d, false, false);
        d = __builtin_amdgcn_wmma_f32_16x16x32_f16(false, a1, false,
              load_b_t(dwa, ct * 16, 32, lane), (short)0, d, false, false);
        const int n = ct * 16 + l15;
#pragma unroll
        for (int j = 0; j < 8; ++j) {
          const int m = ort * 16 + j + 8 * lh;
          const float v = (d[j] + p.attn_pb[br][m]) * gate[n * C_ + m];
          oi[n * C_ + m] = (_Float16)v;
        }
      }
    }
    __syncthreads();

    // GEMM3 (accumulated): acc += pl_w[:, br*64:(br+1)*64] @ oi
    {
      const v16h a0 = load_a_f32(p.pl_w + br * 64, 256, ort * 16, 0, lane);
      const v16h a1 = load_a_f32(p.pl_w + br * 64, 256, ort * 16, 32, lane);
#pragma unroll
      for (int cc = 0; cc < 2; ++cc) {
        const int ct = oc0 + cc;
        acc[cc] = __builtin_amdgcn_wmma_f32_16x16x32_f16(false, a0, false,
                    load_b_t(oi, ct * 16, 0, lane), (short)0, acc[cc], false, false);
        acc[cc] = __builtin_amdgcn_wmma_f32_16x16x32_f16(false, a1, false,
                    load_b_t(oi, ct * 16, 32, lane), (short)0, acc[cc], false, false);
      }
    }
    __syncthreads();                      // before next branch overwrites dwa/gate/oi
  }

  // ---- epilogue: (acc + pl_b) * scale + residual ----
#pragma unroll
  for (int cc = 0; cc < 2; ++cc) {
    const int t = (oc0 + cc) * 16 + l15;
#pragma unroll
    for (int j = 0; j < 8; ++j) {
      const int m = ort * 16 + j + 8 * lh;
      const float v = (acc[cc][j] + p.pl_b[m]) * p.scale[m] + xs[m * TP + HALO + t];
      p.out[(size_t)b * C_ * TT + (size_t)m * TT + (t0 + t)] = v;
    }
  }
}

extern "C" void kernel_launch(void* const* d_in, const int* in_sizes, int n_in,
                              void* d_out, int out_size, void* d_ws, size_t ws_size,
                              hipStream_t stream) {
  (void)in_sizes; (void)n_in; (void)out_size; (void)d_ws; (void)ws_size;
  Params p;
  p.x    = (const float*)d_in[0];
  p.ln_w = (const float*)d_in[1];
  p.ln_b = (const float*)d_in[2];
  p.pf_w = (const float*)d_in[3];
  p.pf_b = (const float*)d_in[4];
  p.pl_w = (const float*)d_in[5];
  p.pl_b = (const float*)d_in[6];
  p.scale = (const float*)d_in[7];
  for (int i = 0; i < 4; ++i) {
    p.attn_dw[i] = (const float*)d_in[8 + i];
    p.attn_db[i] = (const float*)d_in[12 + i];
    p.attn_pw[i] = (const float*)d_in[16 + i];
    p.attn_pb[i] = (const float*)d_in[20 + i];
    p.conv_dw[i] = (const float*)d_in[24 + i];
    p.conv_db[i] = (const float*)d_in[28 + i];
  }
  p.out = (float*)d_out;

  const size_t smem_bytes = 118784;           // see LDS carve-out in kernel
  const dim3 grid(NB * (TT / TILE));          // 8 * 750 = 6000 workgroups
  const dim3 block(256);                      // 8 wave32s
  fused_block_kernel<<<grid, block, smem_bytes, stream>>>(p);
}
